// GAT_58643483460245
// MI455X (gfx1250) — compile-verified, matrix operand order checked by reading, not run on hardware
//
#include <hip/hip_runtime.h>

// ---------------------------------------------------------------- constants
#define NN      100000      // nodes
#define NE      3200000     // edges (without self loops)
#define EPALL   (NE + NN)   // edges + self loops
#define HID     64
#define NHEAD   4
#define NG      1024
#define NEGSL   0.2f

typedef __attribute__((ext_vector_type(16))) __bf16 v16bf;
typedef __attribute__((ext_vector_type(8)))  float  v8f;

// float atomic max via int/uint ordering trick (init with -inf)
__device__ __forceinline__ void atomicMaxF(float* addr, float v) {
  if (v >= 0.0f) atomicMax((int*)addr, __float_as_int(v));
  else           atomicMin((unsigned int*)addr, __float_as_uint(v));
}

__device__ __forceinline__ void edge_sd(const int* __restrict__ ei, int i, int& s, int& d) {
  if (i < NE) { s = ei[i]; d = ei[NE + i]; }
  else        { s = i - NE; d = i - NE; }
  s = min(max(s, 0), NN - 1);
  d = min(max(d, 0), NN - 1);
}

// ---------------------------------------------------- weight prep (one-shot)
// BT[col][k] = bf16(W[k][col]) : fragment-ready, per-lane contiguous.
__global__ void k_prep_w(const float* __restrict__ W, __bf16* __restrict__ BT, int K) {
  int t = blockIdx.x * blockDim.x + threadIdx.x;
  if (t >= K * HID) return;
  int col = t / K, k = t - col * K;
  BT[t] = (__bf16)W[(size_t)k * HID + col];   // native v_cvt to bf16
}

// ------------------------------------------------------- WMMA GEMM, N == 64
// Out[M,64] = A[M,K] @ W[K,64] (+ bias) (+ relu); one wave per 16-row tile.
// A is fp32 (converted to bf16 in-register), BT is prepped bf16 [64][K].
template<int K, bool RELU>
__global__ __launch_bounds__(256)
void wmma_gemm64(const float* __restrict__ A, const __bf16* __restrict__ BT,
                 const float* __restrict__ bias, float* __restrict__ Out, int M) {
  int wave  = (blockIdx.x * blockDim.x + threadIdx.x) >> 5;
  int lane  = threadIdx.x & 31;
  int tile0 = wave * 16;
  if (tile0 >= M) return;                 // wave-uniform: EXEC stays all-ones
  int half = lane >> 4;                   // 0 or 1
  int mr   = lane & 15;                   // A-row / B-col / D-col index
  int arow = tile0 + mr;
  if (arow >= M) arow = M - 1;

  v8f acc[4] = {};
  #pragma unroll
  for (int ks = 0; ks < K; ks += 32) {
    int kb = ks + half * 8;
    // A fragment: row 'arow', K = kb+{0..7, 16..23}; fp32 -> bf16 in-register
    const float* ap = A + (size_t)arow * K + kb;
    v16bf a;
    #pragma unroll
    for (int i = 0; i < 8; ++i) {
      a[i]     = (__bf16)ap[i];
      a[8 + i] = (__bf16)ap[16 + i];
    }
    #pragma unroll
    for (int nt = 0; nt < 4; ++nt) {
      // B fragment: col nt*16+mr, same K pattern; contiguous 2x16B per lane
      const __bf16* bp = BT + (size_t)(nt * 16 + mr) * K + kb;
      v16bf b;
      #pragma unroll
      for (int i = 0; i < 8; ++i) { b[i] = bp[i]; b[8 + i] = bp[16 + i]; }
      acc[nt] = __builtin_amdgcn_wmma_f32_16x16x32_bf16(
                    false, a, false, b, (short)0, acc[nt], false, false);
    }
  }

  int drow0 = tile0 + half * 8;           // D: lane holds rows drow0..+7, col nt*16+mr
  #pragma unroll
  for (int nt = 0; nt < 4; ++nt) {
    int col  = nt * 16 + mr;
    float bv = bias ? bias[col] : 0.0f;
    #pragma unroll
    for (int v = 0; v < 8; ++v) {
      int r = drow0 + v;
      if (r < M) {
        float val = acc[nt][v] + bv;
        if (RELU) val = fmaxf(val, 0.0f);
        Out[(size_t)r * HID + col] = val;
      }
    }
  }
}

// ----------------------------------------------------------- layer kernels
__global__ void k_init_layer(float* __restrict__ hnext, float* __restrict__ nmax,
                             float* __restrict__ nsum) {
  int t = blockIdx.x * blockDim.x + threadIdx.x;
  if (t >= NN * HID) return;
  hnext[t] = 0.0f;
  if (t < NN * NHEAD) {
    nmax[t] = __uint_as_float(0xFF800000u);   // -inf
    nsum[t] = 0.0f;
  }
}

// per (node, head): alpha_src / alpha_dst dot products
__global__ void k_alpha(const float* __restrict__ xw, const float* __restrict__ att_s,
                        const float* __restrict__ att_d, float* __restrict__ as,
                        float* __restrict__ ad) {
  int t = blockIdx.x * blockDim.x + threadIdx.x;
  if (t >= NN * NHEAD) return;
  int n = t >> 2, h = t & 3;
  const float* xr = xw + (size_t)n * HID + h * 16;
  const float* sv = att_s + h * 16;
  const float* dv = att_d + h * 16;
  float ss = 0.0f, sd = 0.0f;
  #pragma unroll
  for (int j = 0; j < 16; ++j) { ss += xr[j] * sv[j]; sd += xr[j] * dv[j]; }
  as[t] = ss;
  ad[t] = sd;
}

__global__ void k_edge_max(const int* __restrict__ ei, const float* __restrict__ as,
                           const float* __restrict__ ad, float* __restrict__ nmax) {
  int i = blockIdx.x * blockDim.x + threadIdx.x;
  if (i >= EPALL) return;
  int s, d; edge_sd(ei, i, s, d);
  float4 a = ((const float4*)as)[s];
  float4 b = ((const float4*)ad)[d];
  float e0 = a.x + b.x, e1 = a.y + b.y, e2 = a.z + b.z, e3 = a.w + b.w;
  e0 = e0 > 0.0f ? e0 : NEGSL * e0;
  e1 = e1 > 0.0f ? e1 : NEGSL * e1;
  e2 = e2 > 0.0f ? e2 : NEGSL * e2;
  e3 = e3 > 0.0f ? e3 : NEGSL * e3;
  float* m = nmax + (size_t)d * 4;
  atomicMaxF(m + 0, e0); atomicMaxF(m + 1, e1);
  atomicMaxF(m + 2, e2); atomicMaxF(m + 3, e3);
}

__global__ void k_edge_sum(const int* __restrict__ ei, const float* __restrict__ as,
                           const float* __restrict__ ad, const float* __restrict__ nmax,
                           float* __restrict__ nsum) {
  int i = blockIdx.x * blockDim.x + threadIdx.x;
  if (i >= EPALL) return;
  int s, d; edge_sd(ei, i, s, d);
  float4 a = ((const float4*)as)[s];
  float4 b = ((const float4*)ad)[d];
  float4 m = ((const float4*)nmax)[d];
  float e0 = a.x + b.x, e1 = a.y + b.y, e2 = a.z + b.z, e3 = a.w + b.w;
  e0 = e0 > 0.0f ? e0 : NEGSL * e0;
  e1 = e1 > 0.0f ? e1 : NEGSL * e1;
  e2 = e2 > 0.0f ? e2 : NEGSL * e2;
  e3 = e3 > 0.0f ? e3 : NEGSL * e3;
  float* sp = nsum + (size_t)d * 4;
  atomicAdd(sp + 0, __expf(e0 - m.x));
  atomicAdd(sp + 1, __expf(e1 - m.y));
  atomicAdd(sp + 2, __expf(e2 - m.z));
  atomicAdd(sp + 3, __expf(e3 - m.w));
}

// one thread per (edge, 8-channel chunk): 8 contiguous loads + 8 atomic adds
__global__ void k_edge_msg(const int* __restrict__ ei, const float* __restrict__ as,
                           const float* __restrict__ ad, const float* __restrict__ nmax,
                           const float* __restrict__ nsum, const float* __restrict__ xw,
                           float* __restrict__ out) {
  unsigned t = blockIdx.x * blockDim.x + threadIdx.x;
  if (t >= (unsigned)EPALL * 8u) return;
  int i  = (int)(t >> 3);
  int c8 = (int)(t & 7u);
  int h  = c8 >> 1;
  int s, d; edge_sd(ei, i, s, d);
  float e = as[(size_t)s * 4 + h] + ad[(size_t)d * 4 + h];
  e = e > 0.0f ? e : NEGSL * e;
  float alpha = __expf(e - nmax[(size_t)d * 4 + h]) /
                (nsum[(size_t)d * 4 + h] + 1e-16f);
  const float* xs = xw  + (size_t)s * HID + c8 * 8;
  float*       ob = out + (size_t)d * HID + c8 * 8;
  #pragma unroll
  for (int j = 0; j < 8; ++j) atomicAdd(&ob[j], xs[j] * alpha);
}

__global__ void k_finalize(float* __restrict__ h, const float* __restrict__ bg) {
  int t = blockIdx.x * blockDim.x + threadIdx.x;
  if (t >= NN * HID) return;
  int c = t & (HID - 1);
  h[t] = fmaxf(h[t] + bg[c], 0.0f);
}

// ------------------------------------------------------------ pooling + MLP
__global__ void k_pool_zero(float* __restrict__ gsum, float* __restrict__ gcnt) {
  int t = blockIdx.x * blockDim.x + threadIdx.x;
  if (t >= NG * HID) return;
  gsum[t] = 0.0f;
  if (t < NG) gcnt[t] = 0.0f;
}

__global__ void k_pool(const float* __restrict__ h, const int* __restrict__ batch,
                       float* __restrict__ gsum, float* __restrict__ gcnt) {
  int t = blockIdx.x * blockDim.x + threadIdx.x;
  if (t >= NN * HID) return;
  int n = t >> 6, c = t & (HID - 1);
  int b = min(max(batch[n], 0), NG - 1);
  atomicAdd(&gsum[(size_t)b * HID + c], h[t]);
  if (c == 0) atomicAdd(&gcnt[b], 1.0f);
}

__global__ void k_gfeat(const float* __restrict__ gsum, const float* __restrict__ gcnt,
                        float* __restrict__ gfeat) {
  int t = blockIdx.x * blockDim.x + threadIdx.x;
  if (t >= NG * 2 * HID) return;
  int g = t >> 7, c = t & 127;
  float v;
  if (c < HID) v = gsum[(size_t)g * HID + c];
  else         v = gsum[(size_t)g * HID + (c - HID)] / fmaxf(gcnt[g], 1.0f);
  gfeat[t] = v;
}

__global__ void k_mlp2(const float* __restrict__ ghid, const float* __restrict__ W2,
                       const float* __restrict__ b2, float* __restrict__ out) {
  int g = blockIdx.x * blockDim.x + threadIdx.x;
  if (g >= NG) return;
  const float* r = ghid + (size_t)g * HID;
  float acc = 0.0f;
  #pragma unroll
  for (int c = 0; c < HID; ++c) acc += r[c] * W2[c];
  out[g] = acc + b2[0];
}

// ---------------------------------------------------------------- dispatch
extern "C" void kernel_launch(void* const* d_in, const int* in_sizes, int n_in,
                              void* d_out, int out_size, void* d_ws, size_t ws_size,
                              hipStream_t stream) {
  const float* x       = (const float*)d_in[0];
  const int*   ei      = (const int*)  d_in[1];
  // d_in[2] edge_attr: unused by the reference model
  const int*   batch   = (const int*)  d_in[3];
  const float* W_embed = (const float*)d_in[4];
  const float* b_embed = (const float*)d_in[5];
  const float* W_gat   = (const float*)d_in[6];   // [3,64,64]
  const float* att_src = (const float*)d_in[7];   // [3,4,16]
  const float* att_dst = (const float*)d_in[8];   // [3,4,16]
  const float* b_gat   = (const float*)d_in[9];   // [3,64]
  const float* W_h1    = (const float*)d_in[10];  // [128,64]
  const float* b_h1    = (const float*)d_in[11];
  const float* W_h2    = (const float*)d_in[12];  // [64,1]
  const float* b_h2    = (const float*)d_in[13];

  float* ws    = (float*)d_ws;
  float* hA    = ws;                               // [NN,64]
  float* hB    = hA    + (size_t)NN * HID;         // [NN,64]
  float* xw    = hB    + (size_t)NN * HID;         // [NN,64]
  float* as    = xw    + (size_t)NN * HID;         // [NN,4]
  float* ad    = as    + (size_t)NN * NHEAD;       // [NN,4]
  float* nmax  = ad    + (size_t)NN * NHEAD;       // [NN,4]
  float* nsum  = nmax  + (size_t)NN * NHEAD;       // [NN,4]
  float* gsum  = nsum  + (size_t)NN * NHEAD;       // [NG,64]
  float* gcnt  = gsum  + (size_t)NG * HID;         // [NG]
  float* gfeat = gcnt  + NG;                       // [NG,128]
  float* ghid  = gfeat + (size_t)NG * 2 * HID;     // [NG,64]
  // bf16 weight area (16B-aligned: float area is a multiple of 16 bytes)
  __bf16* bfw   = (__bf16*)(ghid + (size_t)NG * HID);
  __bf16* WembT = bfw;                              // [64][128]
  __bf16* WgatT = WembT + (size_t)HID * 128;        // 3 x [64][64]
  __bf16* Wh1T  = WgatT + (size_t)3 * HID * HID;    // [64][128]

  const int TB = 256;
  auto cdiv = [](long a, long b) { return (int)((a + b - 1) / b); };
  const int tilesN = cdiv(NN, 16);    // 6250 (exact)
  const int wpb    = TB / 32;         // waves per block

  // prep: transpose + bf16-convert all GEMM weights (fragment-ready layout)
  k_prep_w<<<cdiv(128 * HID, TB), TB, 0, stream>>>(W_embed, WembT, 128);
  for (int l = 0; l < 3; ++l)
    k_prep_w<<<cdiv(HID * HID, TB), TB, 0, stream>>>(
        W_gat + (size_t)l * HID * HID, WgatT + (size_t)l * HID * HID, HID);
  k_prep_w<<<cdiv(128 * HID, TB), TB, 0, stream>>>(W_h1, Wh1T, 128);

  // embed: hA = x @ W_embed + b_embed
  wmma_gemm64<128, false><<<cdiv(tilesN, wpb), TB, 0, stream>>>(x, WembT, b_embed, hA, NN);

  float* hcur = hA;
  float* hnext = hB;
  for (int l = 0; l < 3; ++l) {
    k_init_layer<<<cdiv((long)NN * HID, TB), TB, 0, stream>>>(hnext, nmax, nsum);
    wmma_gemm64<64, false><<<cdiv(tilesN, wpb), TB, 0, stream>>>(
        hcur, WgatT + (size_t)l * HID * HID, nullptr, xw, NN);
    k_alpha<<<cdiv((long)NN * NHEAD, TB), TB, 0, stream>>>(
        xw, att_src + (size_t)l * HID, att_dst + (size_t)l * HID, as, ad);
    k_edge_max<<<cdiv(EPALL, TB), TB, 0, stream>>>(ei, as, ad, nmax);
    k_edge_sum<<<cdiv(EPALL, TB), TB, 0, stream>>>(ei, as, ad, nmax, nsum);
    k_edge_msg<<<cdiv((long)EPALL * 8, TB), TB, 0, stream>>>(ei, as, ad, nmax, nsum, xw, hnext);
    k_finalize<<<cdiv((long)NN * HID, TB), TB, 0, stream>>>(hnext, b_gat + (size_t)l * HID);
    float* tmp = hcur; hcur = hnext; hnext = tmp;
  }

  // pooling: sum + mean concat
  k_pool_zero<<<cdiv((long)NG * HID, TB), TB, 0, stream>>>(gsum, gcnt);
  k_pool<<<cdiv((long)NN * HID, TB), TB, 0, stream>>>(hcur, batch, gsum, gcnt);
  k_gfeat<<<cdiv((long)NG * 2 * HID, TB), TB, 0, stream>>>(gsum, gcnt, gfeat);

  // head MLP: relu(g @ W_h1 + b_h1) @ W_h2 + b_h2
  wmma_gemm64<128, true><<<cdiv((long)cdiv(NG, 16), wpb), TB, 0, stream>>>(
      gfeat, Wh1T, b_h1, ghid, NG);
  k_mlp2<<<cdiv(NG, TB), TB, 0, stream>>>(ghid, W_h2, b_h2, (float*)d_out);
}
